// BiDAF_27144193310907
// MI455X (gfx1250) — compile-verified, matrix operand order checked by reading, not run on hardware
//
#include <hip/hip_runtime.h>
#include <stdint.h>

// ---------------------------------------------------------------------------
// BiDAF on gfx1250: bf16 WMMA GEMMs + persistent-wave LSTM recurrence.
// ---------------------------------------------------------------------------

typedef __attribute__((ext_vector_type(16))) __bf16 v16bf;
typedef __attribute__((ext_vector_type(8)))  float  v8f;

struct U4x2 { uint4 lo, hi; };   // 32 bytes == v16bf

__device__ __forceinline__ uint16_t f2bf(float f) {
  uint32_t u = __builtin_bit_cast(uint32_t, f);
  uint32_t r = u + 0x7FFFu + ((u >> 16) & 1u);
  return (uint16_t)(r >> 16);
}
__device__ __forceinline__ float sigf(float x) { return 1.f / (1.f + __expf(-x)); }

// ---------------------------------------------------------------------------
// Generic batched GEMM: C[b][m][n] = sum_k A[b][m][k] * W[b][n][k] (+ bias[n])
// A,W bf16 row-major with arbitrary leading dims / batch strides (elements).
// One 16x16 C tile per wave; K must be a multiple of 32; M,N multiples of 16.
// ---------------------------------------------------------------------------
__global__ void __launch_bounds__(256) k_gemm(
    const uint16_t* __restrict__ A, int lda, long long sA,
    const uint16_t* __restrict__ W, int ldw, long long sW,
    float* __restrict__ C, long long sC, int ldc,
    const float* __restrict__ bias, int tilesM, int tilesN, int K)
{
  int wv = threadIdx.x >> 5, lane = threadIdx.x & 31;
  int tile = blockIdx.x * 8 + wv;
  if (tile >= tilesM * tilesN) return;              // wave-uniform
  int tm = tile / tilesN, tn = tile % tilesN;
  const uint16_t* Ab = A + (size_t)blockIdx.z * sA;
  const uint16_t* Wb = W + (size_t)blockIdx.z * sW;
  float*           Cb = C + (size_t)blockIdx.z * sC;

  int r  = lane & 15;
  int kb = (lane >> 4) << 3;                        // 0 or 8
  const uint16_t* ap = Ab + (size_t)(tm * 16 + r) * lda + kb;
  const uint16_t* wp = Wb + (size_t)(tn * 16 + r) * ldw + kb;

  v8f acc = {};
  for (int k0 = 0; k0 < K; k0 += 32) {
    U4x2 da, dw;
    da.lo = *(const uint4*)(ap);      da.hi = *(const uint4*)(ap + 16);
    dw.lo = *(const uint4*)(wp);      dw.hi = *(const uint4*)(wp + 16);
    ap += 32; wp += 32;
    v16bf av = __builtin_bit_cast(v16bf, da);
    v16bf bv = __builtin_bit_cast(v16bf, dw);
    acc = __builtin_amdgcn_wmma_f32_16x16x32_bf16(false, av, false, bv,
                                                  (short)0, acc, false, false);
  }
  int n  = tn * 16 + (lane & 15);
  int m0 = tm * 16 + ((lane >> 4) << 3);
  float bb = bias ? bias[n] : 0.f;
#pragma unroll
  for (int v = 0; v < 8; v++) Cb[(size_t)(m0 + v) * ldc + n] = acc[v] + bb;
}

// ---------------------------------------------------------------------------
// Persistent bidirectional LSTM recurrence.
// gates(t) = xg[t] + h(t-1) @ U^T ; U is (1024,256) bf16.
// One block per sequence; 32 waves: wave w owns M-tile (w&1) and h-tile (w>>1)
// for all four gates -> elementwise LSTM math entirely in registers.
// ---------------------------------------------------------------------------
struct RecArg {
  const float*    xg;     // (L*32, 1024) f32
  const uint16_t* U;      // (1024, 256) bf16
  float*          outF;   // (L*32, outStride)
  uint16_t*       outBf;  // optional bf16 mirror
  int L, rev, outStride, outOff;
};
struct RecArgs4 { RecArg a[4]; };

__global__ void __launch_bounds__(1024) k_lstm_rec(RecArgs4 args, int nseq) {
  if ((int)blockIdx.x >= nseq) return;
  RecArg ra = args.a[blockIdx.x];
  __shared__ __align__(16) uint16_t hsh[32 * 256];  // h state, bf16, 16KB

  int tid = threadIdx.x, lane = tid & 31, wv = tid >> 5;
  for (int i = tid; i < 32 * 256; i += 1024) hsh[i] = 0;

  int tmy = wv & 1;                  // M tile (batch rows 0-15 / 16-31)
  int hq  = wv >> 1;                 // h-column tile (16 cols)
  int m0  = tmy * 16 + ((lane >> 4) << 3);
  int hc  = hq * 16 + (lane & 15);
  int ar  = tmy * 16 + (lane & 15);  // A-tile row this lane loads
  int akb = (lane >> 4) << 3;

  float creg[8];
#pragma unroll
  for (int v = 0; v < 8; v++) creg[v] = 0.f;
  __syncthreads();

  for (int s = 0; s < ra.L; s++) {
    int t = ra.rev ? (ra.L - 1 - s) : s;
    v8f acc[4];
    acc[0] = {}; acc[1] = {}; acc[2] = {}; acc[3] = {};

    for (int k0 = 0; k0 < 256; k0 += 32) {
      U4x2 da;
      const uint16_t* hp = &hsh[ar * 256 + k0 + akb];
      da.lo = *(const uint4*)(hp); da.hi = *(const uint4*)(hp + 16);
      v16bf av = __builtin_bit_cast(v16bf, da);
#pragma unroll
      for (int g = 0; g < 4; g++) {
        const uint16_t* up =
            ra.U + (size_t)(g * 256 + hq * 16 + (lane & 15)) * 256 + k0 + akb;
        U4x2 dw;
        dw.lo = *(const uint4*)(up); dw.hi = *(const uint4*)(up + 16);
        v16bf bv = __builtin_bit_cast(v16bf, dw);
        acc[g] = __builtin_amdgcn_wmma_f32_16x16x32_bf16(
            false, av, false, bv, (short)0, acc[g], false, false);
      }
    }

    const float* xr = ra.xg + (size_t)t * 32 * 1024;
    uint16_t hnew[8];
#pragma unroll
    for (int v = 0; v < 8; v++) {
      int m = m0 + v;
      const float* xm = xr + (size_t)m * 1024 + hc;
      float gi = acc[0][v] + xm[0];
      float gf = acc[1][v] + xm[256];
      float gg = acc[2][v] + xm[512];
      float go = acc[3][v] + xm[768];
      float cc = sigf(gf) * creg[v] + sigf(gi) * tanhf(gg);
      creg[v] = cc;
      float h = sigf(go) * tanhf(cc);
      size_t oi = ((size_t)t * 32 + m) * ra.outStride + ra.outOff + hc;
      ra.outF[oi] = h;
      if (ra.outBf) ra.outBf[oi] = f2bf(h);
      hnew[v] = f2bf(h);
    }
    __syncthreads();                 // all reads of hsh done
#pragma unroll
    for (int v = 0; v < 8; v++) hsh[(m0 + v) * 256 + hc] = hnew[v];
    __syncthreads();                 // hsh ready for next step
  }
}

// ---------------------------------------------------------------------------
// Small helper kernels
// ---------------------------------------------------------------------------
__global__ void k_f2bf(const float* __restrict__ s, uint16_t* __restrict__ d,
                       long long n) {
  long long i = (long long)blockIdx.x * 256 + threadIdx.x;
  if (i < n) d[i] = f2bf(s[i]);
}

// (B,L,256) f32 -> time-major (t*32+b, 256) bf16
__global__ void k_in_tp(const float* __restrict__ s, uint16_t* __restrict__ d,
                        int L) {
  long long n = (long long)L * 32 * 256;
  long long i = (long long)blockIdx.x * 256 + threadIdx.x;
  if (i >= n) return;
  int dd = (int)(i & 255);
  long long r = i >> 8;
  int t = (int)(r >> 5), b = (int)(r & 31);
  d[i] = f2bf(s[((size_t)b * L + t) * 256 + dd]);
}

// per-row dot product (one wave per row)
__global__ void __launch_bounds__(256) k_rowdot(const float* __restrict__ X,
                                                int ld, int D,
                                                const float* __restrict__ w,
                                                float* __restrict__ out, int R) {
  int lane = threadIdx.x & 31, wv = threadIdx.x >> 5;
  int r = blockIdx.x * 8 + wv;
  if (r >= R) return;
  const float* xr = X + (size_t)r * ld;
  float a = 0.f;
  for (int d = lane; d < D; d += 32) a += xr[d] * w[d];
  for (int m = 16; m > 0; m >>= 1) a += __shfl_xor(a, m, 32);
  if (!lane) out[r] = a;
}

__global__ void k_cwcq(const float* __restrict__ c, const float* __restrict__ wcq,
                       uint16_t* __restrict__ out, long long n) {
  long long i = (long long)blockIdx.x * 256 + threadIdx.x;
  if (i < n) out[i] = f2bf(c[i] * wcq[i & 511]);
}

// softmax over j (48) of S row; emits bf16 attention (padded to 64) and row max
__global__ void __launch_bounds__(256) k_att_smj(
    const float* __restrict__ Sraw, const float* __restrict__ cw,
    const float* __restrict__ qw, const float* __restrict__ bc,
    const float* __restrict__ bq, const float* __restrict__ bcq,
    uint16_t* __restrict__ aH, float* __restrict__ Smax) {
  int lane = threadIdx.x & 31, wv = threadIdx.x >> 5;
  int row = blockIdx.x * 8 + wv;            // row = b*400 + i
  if (row >= 12800) return;
  int b = row / 400, i = row % 400;
  float bias3 = bc[0] + bq[0] + bcq[0];
  float cwv = cw[i * 32 + b];
  const float* sr = Sraw + (size_t)row * 48;
  float v0 = sr[lane] + qw[lane * 32 + b] + cwv + bias3;
  float v1 = (lane < 16) ? (sr[32 + lane] + qw[(32 + lane) * 32 + b] + cwv + bias3)
                         : -3.4e38f;
  float mx = fmaxf(v0, v1);
  for (int m = 16; m > 0; m >>= 1) mx = fmaxf(mx, __shfl_xor(mx, m, 32));
  float e0 = __expf(v0 - mx);
  float e1 = (lane < 16) ? __expf(v1 - mx) : 0.f;
  float sm = e0 + e1;
  for (int m = 16; m > 0; m >>= 1) sm += __shfl_xor(sm, m, 32);
  uint16_t* arow = aH + (size_t)row * 64;
  arow[lane]      = f2bf(e0 / sm);
  arow[32 + lane] = (lane < 16) ? f2bf(e1 / sm) : (uint16_t)0;  // zero pad 48..63
  if (!lane) Smax[row] = mx;
}

// softmax over i (400) of Smax -> b_att
__global__ void __launch_bounds__(512) k_batt(const float* __restrict__ Smax,
                                              float* __restrict__ batt) {
  __shared__ float sh[16];
  int b = blockIdx.x, tid = threadIdx.x, lane = tid & 31, wv = tid >> 5;
  const float* s = Smax + (size_t)b * 400;
  float mx = -3.4e38f;
  for (int i = tid; i < 400; i += 512) mx = fmaxf(mx, s[i]);
  for (int m = 16; m > 0; m >>= 1) mx = fmaxf(mx, __shfl_xor(mx, m, 32));
  if (!lane) sh[wv] = mx;
  __syncthreads();
  if (tid < 16) {
    float v = sh[tid];
    for (int m = 8; m > 0; m >>= 1) v = fmaxf(v, __shfl_xor(v, m, 16));
    if (!tid) sh[0] = v;
  }
  __syncthreads();
  float MX = sh[0];
  __syncthreads();
  float sm = 0.f;
  for (int i = tid; i < 400; i += 512) sm += __expf(s[i] - MX);
  for (int m = 16; m > 0; m >>= 1) sm += __shfl_xor(sm, m, 32);
  if (!lane) sh[wv] = sm;
  __syncthreads();
  if (tid < 16) {
    float v = sh[tid];
    for (int m = 8; m > 0; m >>= 1) v += __shfl_xor(v, m, 16);
    if (!tid) sh[0] = v;
  }
  __syncthreads();
  float SM = sh[0];
  for (int i = tid; i < 400; i += 512) batt[b * 400 + i] = __expf(s[i] - MX) / SM;
}

// q2c[b][d] = sum_i batt[b,i] * c[(i*32+b)*512+d]
__global__ void __launch_bounds__(512) k_q2c(const float* __restrict__ batt,
                                             const float* __restrict__ c,
                                             float* __restrict__ q2c) {
  int b = blockIdx.x, d = threadIdx.x;
  float a = 0.f;
  for (int i = 0; i < 400; i++)
    a += batt[b * 400 + i] * c[((size_t)i * 32 + b) * 512 + d];
  q2c[b * 512 + d] = a;
}

// qT_bf[b][d][jp] (jp padded to 64) from time-major q f32
__global__ void k_qT(const float* __restrict__ q, uint16_t* __restrict__ qT) {
  long long i = (long long)blockIdx.x * 256 + threadIdx.x;
  if (i >= 32LL * 512 * 64) return;
  int jp = (int)(i & 63);
  long long r = i >> 6;
  int d = (int)(r & 511), b = (int)(r >> 9);
  qT[i] = (jp < 48) ? f2bf(q[((size_t)jp * 32 + b) * 512 + d]) : (uint16_t)0;
}

// g = [c, c2q, c*c2q, c*q2c] time-major (12800, 2048), f32 + bf16
__global__ void k_build_g(const float* __restrict__ c, const float* __restrict__ c2q,
                          const float* __restrict__ q2c, float* __restrict__ gF,
                          uint16_t* __restrict__ gH) {
  long long i = (long long)blockIdx.x * 256 + threadIdx.x;
  if (i >= 12800LL * 2048) return;
  long long r = i >> 11;
  int d2 = (int)(i & 2047), sect = d2 >> 9, d = d2 & 511, b = (int)(r & 31);
  float cv = c[r * 512 + d];
  float val;
  if (sect == 0)      val = cv;
  else if (sect == 1) val = c2q[r * 512 + d];
  else if (sect == 2) val = cv * c2q[r * 512 + d];
  else                val = cv * q2c[b * 512 + d];
  gF[i] = val;
  gH[i] = f2bf(val);
}

// p1/p2 logits: one wave per time-major row
__global__ void __launch_bounds__(256) k_plogits(
    const float* __restrict__ g, const float* __restrict__ m,
    const float* __restrict__ m2, const float* __restrict__ p1gw,
    const float* __restrict__ p1gb, const float* __restrict__ p1mw,
    const float* __restrict__ p1mb, const float* __restrict__ p2gw,
    const float* __restrict__ p2gb, const float* __restrict__ p2mw,
    const float* __restrict__ p2mb, float* __restrict__ l1,
    float* __restrict__ l2) {
  int lane = threadIdx.x & 31, wv = threadIdx.x >> 5;
  int row = blockIdx.x * 8 + wv;
  if (row >= 12800) return;
  const float* gr  = g  + (size_t)row * 2048;
  const float* mr  = m  + (size_t)row * 512;
  const float* m2r = m2 + (size_t)row * 512;
  float a1 = 0.f, a2 = 0.f;
  for (int d = lane; d < 2048; d += 32) {
    float gv = gr[d];
    a1 += gv * p1gw[d];
    a2 += gv * p2gw[d];
  }
  for (int d = lane; d < 512; d += 32) {
    a1 += mr[d] * p1mw[d];
    a2 += m2r[d] * p2mw[d];
  }
  for (int s = 16; s > 0; s >>= 1) {
    a1 += __shfl_xor(a1, s, 32);
    a2 += __shfl_xor(a2, s, 32);
  }
  if (!lane) {
    int i = row >> 5, b = row & 31;
    l1[b * 400 + i] = a1 + p1gb[0] + p1mb[0];
    l2[b * 400 + i] = a2 + p2gb[0] + p2mb[0];
  }
}

// final softmax over i per (output, batch)
__global__ void __launch_bounds__(512) k_outsm(const float* __restrict__ l1,
                                               const float* __restrict__ l2,
                                               float* __restrict__ out) {
  __shared__ float sh[16];
  int o = blockIdx.x >> 5, b = blockIdx.x & 31;
  int tid = threadIdx.x, lane = tid & 31, wv = tid >> 5;
  const float* s = (o == 0 ? l1 : l2) + (size_t)b * 400;
  float mx = -3.4e38f;
  for (int i = tid; i < 400; i += 512) mx = fmaxf(mx, s[i]);
  for (int m = 16; m > 0; m >>= 1) mx = fmaxf(mx, __shfl_xor(mx, m, 32));
  if (!lane) sh[wv] = mx;
  __syncthreads();
  if (tid < 16) {
    float v = sh[tid];
    for (int m = 8; m > 0; m >>= 1) v = fmaxf(v, __shfl_xor(v, m, 16));
    if (!tid) sh[0] = v;
  }
  __syncthreads();
  float MX = sh[0];
  __syncthreads();
  float sm = 0.f;
  for (int i = tid; i < 400; i += 512) sm += __expf(s[i] - MX);
  for (int m = 16; m > 0; m >>= 1) sm += __shfl_xor(sm, m, 32);
  if (!lane) sh[wv] = sm;
  __syncthreads();
  if (tid < 16) {
    float v = sh[tid];
    for (int m = 8; m > 0; m >>= 1) v += __shfl_xor(v, m, 16);
    if (!tid) sh[0] = v;
  }
  __syncthreads();
  float SM = sh[0];
  for (int i = tid; i < 400; i += 512)
    out[(size_t)o * 12800 + b * 400 + i] = __expf(s[i] - MX) / SM;
}

// ---------------------------------------------------------------------------
// Host orchestration
// ---------------------------------------------------------------------------
extern "C" void kernel_launch(void* const* d_in, const int* in_sizes, int n_in,
                              void* d_out, int out_size, void* d_ws, size_t ws_size,
                              hipStream_t stream) {
  (void)in_sizes; (void)n_in; (void)out_size; (void)ws_size;
  const float* ctx_in = (const float*)d_in[0];
  const float* qry_in = (const float*)d_in[1];
  const float *W_[4][2][2], *U_[4][2][2], *b_[4][2][2];
  int ii = 2;
  for (int s = 0; s < 4; s++)
    for (int l = 0; l < 2; l++)
      for (int dr = 0; dr < 2; dr++) {
        W_[s][l][dr] = (const float*)d_in[ii++];
        U_[s][l][dr] = (const float*)d_in[ii++];
        b_[s][l][dr] = (const float*)d_in[ii++];
      }
  const float* att_c_w  = (const float*)d_in[50]; const float* att_c_b  = (const float*)d_in[51];
  const float* att_q_w  = (const float*)d_in[52]; const float* att_q_b  = (const float*)d_in[53];
  const float* att_cq_w = (const float*)d_in[54]; const float* att_cq_b = (const float*)d_in[55];
  const float* p1g_w = (const float*)d_in[56]; const float* p1g_b = (const float*)d_in[57];
  const float* p1m_w = (const float*)d_in[58]; const float* p1m_b = (const float*)d_in[59];
  const float* p2g_w = (const float*)d_in[60]; const float* p2g_b = (const float*)d_in[61];
  const float* p2m_w = (const float*)d_in[62]; const float* p2m_b = (const float*)d_in[63];

  char* base = (char*)d_ws; size_t off = 0;
  auto alloc  = [&](size_t bytes) { void* p = base + off; off = (off + bytes + 255) & ~(size_t)255; return p; };
  auto allocF = [&](size_t n) { return (float*)alloc(n * 4); };
  auto allocH = [&](size_t n) { return (uint16_t*)alloc(n * 2); };

  // --- weight conversion to bf16 ---
  const int dins[4][2] = {{256, 512}, {2048, 512}, {512, 512}, {512, 512}};
  uint16_t *Wbf[4][2][2], *Ubf[4][2][2];
  for (int s = 0; s < 4; s++)
    for (int l = 0; l < 2; l++)
      for (int dr = 0; dr < 2; dr++) {
        long long nw = 1024LL * dins[s][l];
        Wbf[s][l][dr] = allocH((size_t)nw);
        Ubf[s][l][dr] = allocH(1024 * 256);
        k_f2bf<<<(unsigned)((nw + 255) / 256), 256, 0, stream>>>(W_[s][l][dr], Wbf[s][l][dr], nw);
        k_f2bf<<<1024, 256, 0, stream>>>(U_[s][l][dr], Ubf[s][l][dr], 262144LL);
      }

  const int RC = 12800, RQ = 1536;  // L*B rows for context / query
  uint16_t* X0c = allocH((size_t)RC * 256);
  uint16_t* X0q = allocH((size_t)RQ * 256);
  k_in_tp<<<(RC * 256 + 255) / 256, 256, 0, stream>>>(ctx_in, X0c, 400);
  k_in_tp<<<(RQ * 256 + 255) / 256, 256, 0, stream>>>(qry_in, X0q, 48);

  float* xgF  = allocF((size_t)RC * 1024);
  float* xgB  = allocF((size_t)RC * 1024);
  float* xgQF = allocF((size_t)RQ * 1024);
  float* xgQB = allocF((size_t)RQ * 1024);
  float* p0F = allocF((size_t)RC * 512); uint16_t* p0H = allocH((size_t)RC * 512);
  float* q0F = allocF((size_t)RQ * 512); uint16_t* q0H = allocH((size_t)RQ * 512);
  float* cF  = allocF((size_t)RC * 512); uint16_t* cH  = allocH((size_t)RC * 512);
  float* qF  = allocF((size_t)RQ * 512); uint16_t* qH  = allocH((size_t)RQ * 512);

  auto gemm = [&](const uint16_t* A, int lda, long long sA,
                  const uint16_t* W, int ldw, long long sW,
                  float* Cf, long long sCf, int ldc, const float* bias,
                  int M, int N, int K, int batch) {
    int tm = M / 16, tn = N / 16, nt = tm * tn;
    dim3 g((unsigned)((nt + 7) / 8), 1, (unsigned)batch);
    k_gemm<<<g, 256, 0, stream>>>(A, lda, sA, W, ldw, sW, Cf, sCf, ldc, bias, tm, tn, K);
  };

  // --- lstm_ctx layer 0 (context + query share weights, 4 sequences) ---
  gemm(X0c, 256, 0, Wbf[0][0][0], 256, 0, xgF,  0, 1024, b_[0][0][0], RC, 1024, 256, 1);
  gemm(X0c, 256, 0, Wbf[0][0][1], 256, 0, xgB,  0, 1024, b_[0][0][1], RC, 1024, 256, 1);
  gemm(X0q, 256, 0, Wbf[0][0][0], 256, 0, xgQF, 0, 1024, b_[0][0][0], RQ, 1024, 256, 1);
  gemm(X0q, 256, 0, Wbf[0][0][1], 256, 0, xgQB, 0, 1024, b_[0][0][1], RQ, 1024, 256, 1);
  {
    RecArgs4 ra{};
    ra.a[0] = {xgF,  Ubf[0][0][0], p0F, p0H, 400, 0, 512, 0};
    ra.a[1] = {xgB,  Ubf[0][0][1], p0F, p0H, 400, 1, 512, 256};
    ra.a[2] = {xgQF, Ubf[0][0][0], q0F, q0H, 48,  0, 512, 0};
    ra.a[3] = {xgQB, Ubf[0][0][1], q0F, q0H, 48,  1, 512, 256};
    k_lstm_rec<<<4, 1024, 0, stream>>>(ra, 4);
  }
  // --- lstm_ctx layer 1 ---
  gemm(p0H, 512, 0, Wbf[0][1][0], 512, 0, xgF,  0, 1024, b_[0][1][0], RC, 1024, 512, 1);
  gemm(p0H, 512, 0, Wbf[0][1][1], 512, 0, xgB,  0, 1024, b_[0][1][1], RC, 1024, 512, 1);
  gemm(q0H, 512, 0, Wbf[0][1][0], 512, 0, xgQF, 0, 1024, b_[0][1][0], RQ, 1024, 512, 1);
  gemm(q0H, 512, 0, Wbf[0][1][1], 512, 0, xgQB, 0, 1024, b_[0][1][1], RQ, 1024, 512, 1);
  {
    RecArgs4 ra{};
    ra.a[0] = {xgF,  Ubf[0][1][0], cF, cH, 400, 0, 512, 0};
    ra.a[1] = {xgB,  Ubf[0][1][1], cF, cH, 400, 1, 512, 256};
    ra.a[2] = {xgQF, Ubf[0][1][0], qF, qH, 48,  0, 512, 0};
    ra.a[3] = {xgQB, Ubf[0][1][1], qF, qH, 48,  1, 512, 256};
    k_lstm_rec<<<4, 1024, 0, stream>>>(ra, 4);
  }

  // --- attention ---
  float* cw = allocF(RC);
  float* qw = allocF(RQ);
  k_rowdot<<<(RC + 7) / 8, 256, 0, stream>>>(cF, 512, 512, att_c_w, cw, RC);
  k_rowdot<<<(RQ + 7) / 8, 256, 0, stream>>>(qF, 512, 512, att_q_w, qw, RQ);
  uint16_t* cwcq = allocH((size_t)RC * 512);
  k_cwcq<<<(RC * 512 + 255) / 256, 256, 0, stream>>>(cF, att_cq_w, cwcq, (long long)RC * 512);
  float* Sraw = allocF(32LL * 400 * 48);
  // S_bilinear[b][i][j]: A=(c*wcq) time-major, W=q time-major (both lda 16384)
  gemm(cwcq, 16384, 512, qH, 16384, 512, Sraw, 400LL * 48, 48, nullptr, 400, 48, 512, 32);
  uint16_t* aH  = allocH(32LL * 400 * 64);
  float* Smax   = allocF(RC);
  k_att_smj<<<(12800 + 7) / 8, 256, 0, stream>>>(Sraw, cw, qw, att_c_b, att_q_b,
                                                 att_cq_b, aH, Smax);
  float* batt = allocF(RC);
  k_batt<<<32, 512, 0, stream>>>(Smax, batt);
  float* q2c = allocF(32 * 512);
  k_q2c<<<32, 512, 0, stream>>>(batt, cF, q2c);
  uint16_t* qT = allocH(32LL * 512 * 64);
  k_qT<<<(unsigned)((32LL * 512 * 64 + 255) / 256), 256, 0, stream>>>(qF, qT);
  float* c2q = allocF((size_t)RC * 512);
  // c2q time-major: batch base b*512, ldc 16384
  gemm(aH, 64, 400LL * 64, qT, 64, 512LL * 64, c2q, 512, 16384, nullptr, 400, 512, 64, 32);
  float* gFb = allocF((size_t)RC * 2048);
  uint16_t* gH = allocH((size_t)RC * 2048);
  k_build_g<<<(unsigned)(((long long)RC * 2048 + 255) / 256), 256, 0, stream>>>(
      cF, c2q, q2c, gFb, gH);

  // --- modeling / output LSTM stacks ---
  float* xFv = allocF((size_t)RC * 512); uint16_t* xH = allocH((size_t)RC * 512);
  float* mF  = allocF((size_t)RC * 512); uint16_t* mH = allocH((size_t)RC * 512);
  float* m2F = allocF((size_t)RC * 512);

  auto bilayer = [&](int s, int l, const uint16_t* Xin, int Kd, float* oF,
                     uint16_t* oH) {
    gemm(Xin, Kd, 0, Wbf[s][l][0], Kd, 0, xgF, 0, 1024, b_[s][l][0], RC, 1024, Kd, 1);
    gemm(Xin, Kd, 0, Wbf[s][l][1], Kd, 0, xgB, 0, 1024, b_[s][l][1], RC, 1024, Kd, 1);
    RecArgs4 ra{};
    ra.a[0] = {xgF, Ubf[s][l][0], oF, oH, 400, 0, 512, 0};
    ra.a[1] = {xgB, Ubf[s][l][1], oF, oH, 400, 1, 512, 256};
    k_lstm_rec<<<2, 1024, 0, stream>>>(ra, 2);
  };
  bilayer(1, 0, gH, 2048, p0F, p0H);   // lstm_m1 layer0
  bilayer(1, 1, p0H, 512, xFv, xH);    // lstm_m1 layer1 -> x
  bilayer(2, 0, xH, 512, p0F, p0H);    // lstm_m2 layer0
  bilayer(2, 1, p0H, 512, mF, mH);     // lstm_m2 layer1 -> m
  bilayer(3, 0, mH, 512, p0F, p0H);    // lstm_out layer0
  bilayer(3, 1, p0H, 512, m2F, nullptr); // lstm_out layer1 -> m2 (f32 only)
  (void)xFv;

  // --- output heads ---
  float* l1 = allocF(RC);
  float* l2 = allocF(RC);
  k_plogits<<<(12800 + 7) / 8, 256, 0, stream>>>(gFb, mF, m2F, p1g_w, p1g_b, p1m_w,
                                                 p1m_b, p2g_w, p2g_b, p2m_w, p2m_b,
                                                 l1, l2);
  k_outsm<<<64, 512, 0, stream>>>(l1, l2, (float*)d_out);
}
